// LSTMTagger_40450001994228
// MI455X (gfx1250) — compile-verified, matrix operand order checked by reading, not run on hardware
//
#include <hip/hip_runtime.h>
#include <hip/hip_bf16.h>

// ---------------------------------------------------------------------------
// LSTM tagger for MI455X (gfx1250).
//   S=8192 words, LC=16 chars, CE=CH=64, WE=128, HW=256, TAGS=50
// Pipeline (all on `stream`):
//   k_init       : zero the cross-WGP phase-barrier counter
//   k_char_lstm  : char LSTM, all words in parallel, bf16 WMMA GEMM per step,
//                  weights + embedding table staged into LDS via TDM
//   k_wih_gemm   : Xg = [emb|char_feat] @ word_Wih.T + b  (bf16 WMMA GEMM,
//                  weight tile staged via TDM)
//   k_word_lstm  : 4-WGP cooperative sequential recurrence, Whh.T resident in
//                  LDS (bf16, CDNA5's 320KB/WGP makes this fit), WMMA matvec
//   k_tag        : logits + log_softmax
// Workspace layout (assumes ws_size >= ~43 MB):
//   [0,2MB)    char_feat  f32 [8192][64]
//   [2,34MB)   Xg         f32 [8192][1024]
//   [34,42MB)  Hout       f32 [8192][256]
//   [42MB+)    act        f32 [2][1024], then u32 barrier counter
// ---------------------------------------------------------------------------

#define S_    8192
#define LC_   16
#define CH_   64
#define WE_   128
#define HW_   256
#define KW_   192      // WE + CH
#define TAGS_ 50

typedef __bf16 bf16;
typedef bf16  v16bf __attribute__((ext_vector_type(16)));
typedef bf16  v8bf  __attribute__((ext_vector_type(8)));
typedef float v8f   __attribute__((ext_vector_type(8)));
typedef float v4f   __attribute__((ext_vector_type(4)));

__device__ __forceinline__ float sigmoidf_(float x) { return 1.f / (1.f + __expf(-x)); }

__device__ __forceinline__ v8f zero_v8f() {
  v8f z;
#pragma unroll
  for (int i = 0; i < 8; ++i) z[i] = 0.f;
  return z;
}

// Build a 16x32 bf16 WMMA fragment (A or B role) from an LDS row.
// Layout per ISA 7.12.2: lanes 0-15 hold K={0..7,16..23}, lanes 16-31 hold
// K={8..15,24..31} of row (lane&15). rowk0 points at K=0 of this lane's row
// and must be 16B aligned -> two ds_load_b128.
__device__ __forceinline__ v16bf frag16(const bf16* rowk0, int lane) {
  int half = (lane >> 4) & 1;                       // 0: K base 0, 1: K base 8
  const v8bf* p0 = (const v8bf*)(rowk0 + half * 8);
  const v8bf* p1 = (const v8bf*)(rowk0 + 16 + half * 8);
  v8bf lo = *p0, hi = *p1;
  v16bf r;
#pragma unroll
  for (int i = 0; i < 8; ++i) { r[i] = lo[i]; r[8 + i] = hi[i]; }
  return r;
}

// ---------------------------------------------------------------------------
// Tensor Data Mover: DMA a dense 2D f32 tile (rows x row_elems, row stride
// row_stride_elems) from global memory into LDS at byte offset lds_off.
// Descriptor layout per CDNA5 ISA ch.8 (D# groups 0/1; groups 2/3 zero = 2D).
// ---------------------------------------------------------------------------
#if __has_builtin(__builtin_amdgcn_tensor_load_to_lds) && \
    __has_builtin(__builtin_amdgcn_s_wait_tensorcnt)
#define HAVE_TDM 1
#else
#define HAVE_TDM 0
#endif

#if HAVE_TDM
typedef unsigned int u32x4 __attribute__((ext_vector_type(4)));
typedef int          i32x4 __attribute__((ext_vector_type(4)));
typedef int          i32x8 __attribute__((ext_vector_type(8)));

__device__ __forceinline__ void tdm_load_2d_f32(unsigned lds_off, const void* gsrc,
                                                unsigned rows, unsigned row_elems,
                                                unsigned row_stride_elems) {
  unsigned long long ga = (unsigned long long)(uintptr_t)gsrc;
  u32x4 g0;
  g0[0] = 1u;                                            // count=1 (valid D#)
  g0[1] = lds_off;                                       // lds_addr (bytes)
  g0[2] = (unsigned)(ga & 0xFFFFFFFFu);                  // global_addr[31:0]
  g0[3] = (unsigned)((ga >> 32) & 0x01FFFFFFu)           // global_addr[56:32]
        | (2u << 30);                                    // type=2 ("image")
  i32x8 g1;
  g1[0] = (int)(2u << 16);                               // data_size=2 (4 bytes)
  g1[1] = (int)((row_elems & 0xFFFFu) << 16);            // tensor_dim0[15:0]
  g1[2] = (int)(((row_elems >> 16) & 0xFFFFu)            // tensor_dim0[31:16]
        | ((rows & 0xFFFFu) << 16));                     // tensor_dim1[15:0]
  g1[3] = (int)(((rows >> 16) & 0xFFFFu)                 // tensor_dim1[31:16]
        | ((row_elems & 0xFFFFu) << 16));                // tile_dim0
  g1[4] = (int)(rows & 0xFFFFu);                         // tile_dim1 (tile_dim2=0)
  g1[5] = (int)row_stride_elems;                         // tensor_dim0_stride[31:0]
  g1[6] = 0;                                             // stride hi / dim1_stride
  g1[7] = 0;
  i32x4 g2 = {0, 0, 0, 0};
  i32x4 g3 = {0, 0, 0, 0};
#if defined(__clang_major__) && (__clang_major__ >= 23)
  i32x8 g4 = {0, 0, 0, 0, 0, 0, 0, 0};
  __builtin_amdgcn_tensor_load_to_lds(g0, g1, g2, g3, g4, 0);
#else
  __builtin_amdgcn_tensor_load_to_lds(g0, g1, g2, g3, 0);
#endif
}
#define LDS_OFFSET(p) ((unsigned)(unsigned long long)(uintptr_t)(void*)(p))
#endif  // HAVE_TDM

// ---------------------------------------------------------------------------
__global__ void k_init(unsigned* c) { *c = 0u; }

// ---------------------------------------------------------------------------
// Char LSTM: 128 blocks x 64 words. Per step: G[64][256] = A[64][128]@W[128][256]
// with A = [x_t | h], W.T stored row-per-output (Bt layout) in LDS as bf16.
// LDS strides padded: 136 bf16 = 272B rows (16B aligned, conflict-free).
// The char embedding table (128x64) lives in LDS as bf16 for the whole kernel.
#define KS_C 136
#define AS_C 136
__global__ __launch_bounds__(256) void k_char_lstm(
    const int* __restrict__ char_idxs, const int* __restrict__ char_lens,
    const float* __restrict__ char_emb,
    const float* __restrict__ Wih, const float* __restrict__ Whh,
    const float* __restrict__ bih, const float* __restrict__ bhh,
    float* __restrict__ char_feat) {
  extern __shared__ char smem[];
  bf16*  Wt   = (bf16*)smem;                                   // 256*136*2 = 69632
  bf16*  A    = (bf16*)(smem + 69632);                         // 64*136*2  = 17408
  float* G    = (float*)(smem + 69632 + 17408);                // 64*256*4  = 65536 (doubles as f32 DMA scratch)
  float* bias = (float*)(smem + 69632 + 17408 + 65536);        // 1024
  float* Cst  = (float*)(smem + 69632 + 17408 + 65536 + 1024); // 64*64*4   = 16384
  bf16*  EmbT = (bf16*)(smem + 69632 + 17408 + 65536 + 1024 + 16384); // 128*64*2 = 16384

  const int tid  = threadIdx.x;
  const int lane = tid & 31;
  const int wv   = tid >> 5;
  const int m0   = blockIdx.x * 64;

  if (tid < 256) bias[tid] = bih[tid] + bhh[tid];
  for (int e = tid; e < 64 * 64; e += 256) {
    int m = e >> 6, u = e & 63;
    A[m * AS_C + 64 + u] = (bf16)0.f;   // h0 = 0
    Cst[e] = 0.f;                       // c0 = 0
  }

#if HAVE_TDM
  // TDM the three weight/table tiles through the G scratch, convert to bf16.
  const unsigned g_off = LDS_OFFSET(G);
  if (__builtin_amdgcn_readfirstlane(wv) == 0) {
    tdm_load_2d_f32(g_off, Wih, 256, 64, 64);
    __builtin_amdgcn_s_wait_tensorcnt((short)0);
  }
  __syncthreads();
  for (int e = tid; e < 256 * 64; e += 256) {
    int n = e >> 6, k = e & 63;
    Wt[n * KS_C + k] = (bf16)G[e];
  }
  __syncthreads();
  if (__builtin_amdgcn_readfirstlane(wv) == 0) {
    tdm_load_2d_f32(g_off, Whh, 256, 64, 64);
    __builtin_amdgcn_s_wait_tensorcnt((short)0);
  }
  __syncthreads();
  for (int e = tid; e < 256 * 64; e += 256) {
    int n = e >> 6, k = e & 63;
    Wt[n * KS_C + 64 + k] = (bf16)G[e];
  }
  __syncthreads();
  if (__builtin_amdgcn_readfirstlane(wv) == 0) {
    tdm_load_2d_f32(g_off, char_emb, 128, 64, 64);
    __builtin_amdgcn_s_wait_tensorcnt((short)0);
  }
  __syncthreads();
  for (int e = tid; e < 128 * 64; e += 256) EmbT[e] = (bf16)G[e];
#else
  for (int e = tid; e < 256 * 64; e += 256) {
    int n = e >> 6, k = e & 63;
    Wt[n * KS_C + k]      = (bf16)Wih[n * 64 + k];
    Wt[n * KS_C + 64 + k] = (bf16)Whh[n * 64 + k];
  }
  for (int e = tid; e < 128 * 64; e += 256) EmbT[e] = (bf16)char_emb[e];
#endif

  const int gm     = tid >> 2;          // word row handled in gate phase
  const int gu0    = (tid & 3) << 4;    // hidden-unit base (16 units/thread)
  const int myword = m0 + gm;
  const int mylen  = char_lens[myword];

  for (int t = 0; t < LC_; ++t) {
    __syncthreads();  // G reuse: previous gate phase finished reading G
    // stage x_t: gather rows from LDS-resident bf16 embedding table,
    // 16B (v8bf) chunks -> ds_load_b128/ds_store_b128, 2 chunks per thread
    for (int e = tid; e < 64 * 8; e += 256) {
      int mrow = e >> 3, kq = (e & 7) * 8;
      int ci = char_idxs[(m0 + mrow) * LC_ + t];
      *(v8bf*)&A[mrow * AS_C + kq] = *(const v8bf*)&EmbT[ci * 64 + kq];
    }
    __syncthreads();
    // GEMM: each wave owns M-tile (wv>>1) and 8 N-tiles
    {
      const int mt  = wv >> 1;
      const int ntb = (wv & 1) * 8;
      v8f acc[8];
#pragma unroll
      for (int j = 0; j < 8; ++j) acc[j] = zero_v8f();
#pragma unroll
      for (int kc = 0; kc < 4; ++kc) {
        v16bf af = frag16(&A[(mt * 16 + (lane & 15)) * AS_C + kc * 32], lane);
#pragma unroll
        for (int j = 0; j < 8; ++j) {
          v16bf bfr = frag16(&Wt[((ntb + j) * 16 + (lane & 15)) * KS_C + kc * 32], lane);
          acc[j] = __builtin_amdgcn_wmma_f32_16x16x32_bf16(
              false, af, false, bfr, (short)0, acc[j], false, false);
        }
      }
      // C layout: VGPR r <-> row mt*16 + r + 8*(lane>=16), col = ntile*16 + (lane&15)
#pragma unroll
      for (int j = 0; j < 8; ++j) {
        int n    = (ntb + j) * 16 + (lane & 15);
        int mrow = mt * 16 + ((lane >> 4) << 3);
#pragma unroll
        for (int r = 0; r < 8; ++r) G[(mrow + r) * 256 + n] = acc[j][r];
      }
    }
    __syncthreads();
    // gate phase (PyTorch order i,f,g,o): vectorized b128 LDS traffic.
    {
      bf16  htmp[16];
      float ftmp[16];
      const float* Grow = &G[gm * 256];
#pragma unroll
      for (int q = 0; q < 4; ++q) {
        int u = gu0 + q * 4;
        v4f vi = *(const v4f*)&Grow[u];
        v4f vf = *(const v4f*)&Grow[64 + u];
        v4f vg = *(const v4f*)&Grow[128 + u];
        v4f vo = *(const v4f*)&Grow[192 + u];
        v4f bi = *(const v4f*)&bias[u];
        v4f bfv = *(const v4f*)&bias[64 + u];
        v4f bg = *(const v4f*)&bias[128 + u];
        v4f bo = *(const v4f*)&bias[192 + u];
        v4f vc = *(const v4f*)&Cst[gm * 64 + u];
        v4f cn;
#pragma unroll
        for (int x = 0; x < 4; ++x) {
          float c = sigmoidf_(vf[x] + bfv[x]) * vc[x] +
                    sigmoidf_(vi[x] + bi[x]) * tanhf(vg[x] + bg[x]);
          float h = sigmoidf_(vo[x] + bo[x]) * tanhf(c);
          cn[x] = c;
          htmp[q * 4 + x] = (bf16)h;
          ftmp[q * 4 + x] = h;
        }
        *(v4f*)&Cst[gm * 64 + u] = cn;
      }
      // h feeds next step's A (two b128 stores)
      *(v8bf*)&A[gm * AS_C + 64 + gu0]     = *(v8bf*)&htmp[0];
      *(v8bf*)&A[gm * AS_C + 64 + gu0 + 8] = *(v8bf*)&htmp[8];
      if (t == mylen - 1) {
#pragma unroll
        for (int q = 0; q < 4; ++q)
          *(v4f*)&char_feat[myword * 64 + gu0 + q * 4] = *(v4f*)&ftmp[q * 4];
      }
    }
  }
}

// ---------------------------------------------------------------------------
// Xg = concat(word_emb[idx], char_feat) @ word_Wih.T + bih + bhh
// M=8192, K=192, N=1024. Grid: 128 M-blocks x 16 N-blocks, 64x64 tiles.
// Weight tile DMA'd by TDM into f32 scratch, converted to bf16.
#define AS_B 200
#define BS_B 200
__global__ __launch_bounds__(256) void k_wih_gemm(
    const int* __restrict__ word_idxs, const float* __restrict__ word_emb,
    const float* __restrict__ char_feat, const float* __restrict__ Wih,
    const float* __restrict__ bih, const float* __restrict__ bhh,
    float* __restrict__ Xg) {
  extern __shared__ char smemB[];
  bf16*  Asm  = (bf16*)smemB;                 // 64*200*2 = 25600
  bf16*  Bsm  = (bf16*)(smemB + 25600);       // 64*200*2 = 25600
  float* Bscr = (float*)(smemB + 51200);      // 64*192*4 = 49152 (TDM scratch)
  const int tid  = threadIdx.x;
  const int lane = tid & 31;
  const int wv   = tid >> 5;
  const int m0   = (blockIdx.x & 127) * 64;
  const int n0   = (blockIdx.x >> 7) * 64;

#if HAVE_TDM
  if (__builtin_amdgcn_readfirstlane(wv) == 0) {
    tdm_load_2d_f32(LDS_OFFSET(Bscr), Wih + (size_t)n0 * KW_, 64, KW_, KW_);
    __builtin_amdgcn_s_wait_tensorcnt((short)0);
  }
#endif
  // Stage A rows (gather: x = [emb | char_feat]) while the DMA runs.
  {
    int m = tid >> 2, kq = (tid & 3) * 48;
    int wi = word_idxs[m0 + m];
    for (int k = kq; k < kq + 48; ++k) {
      float v = (k < WE_) ? word_emb[wi * WE_ + k] : char_feat[(m0 + m) * CH_ + (k - WE_)];
      Asm[m * AS_B + k] = (bf16)v;
    }
  }
#if HAVE_TDM
  __syncthreads();  // DMA complete (wave0 waited) + visible to all waves
  {
    int n = tid >> 2, kq = (tid & 3) * 48;
    for (int k = kq; k < kq + 48; ++k)
      Bsm[n * BS_B + k] = (bf16)Bscr[n * KW_ + k];
  }
#else
  {
    int n = tid >> 2, kq = (tid & 3) * 48;
    for (int k = kq; k < kq + 48; ++k)
      Bsm[n * BS_B + k] = (bf16)Wih[(n0 + n) * KW_ + k];
  }
#endif
  __syncthreads();

  const int mt  = wv >> 1;
  const int ntb = (wv & 1) * 2;
  v8f acc[2] = { zero_v8f(), zero_v8f() };
#pragma unroll
  for (int kc = 0; kc < 6; ++kc) {
    v16bf af = frag16(&Asm[(mt * 16 + (lane & 15)) * AS_B + kc * 32], lane);
#pragma unroll
    for (int j = 0; j < 2; ++j) {
      v16bf bfr = frag16(&Bsm[((ntb + j) * 16 + (lane & 15)) * BS_B + kc * 32], lane);
      acc[j] = __builtin_amdgcn_wmma_f32_16x16x32_bf16(
          false, af, false, bfr, (short)0, acc[j], false, false);
    }
  }
#pragma unroll
  for (int j = 0; j < 2; ++j) {
    int n = n0 + (ntb + j) * 16 + (lane & 15);
    float bsum = bih[n] + bhh[n];
#pragma unroll
    for (int r = 0; r < 8; ++r) {
      int mrow = m0 + mt * 16 + ((lane >> 4) << 3) + r;
      Xg[(size_t)mrow * 1024 + n] = acc[j][r] + bsum;
    }
  }
}

// ---------------------------------------------------------------------------
// Sequential word LSTM: 4 blocks, block g owns gate group g (256 outputs).
// Whh.T slice resident in LDS as bf16 (256x264 rows = 132KB; needs CDNA5 LDS),
// staged by two 128-row TDM DMAs through a 128KB f32 scratch.
// Per step: M=1-padded-to-16 WMMA matvec, publish activated gates through L2,
// global-atomic phase barrier, redundant h/c update per block (no h broadcast).
#define BS_W 264
__global__ __launch_bounds__(256) void k_word_lstm(
    const float* __restrict__ Whh, const float* __restrict__ Xg,
    float* __restrict__ act, unsigned* __restrict__ counter,
    float* __restrict__ Hout) {
  extern __shared__ char smem[];
  bf16*  Bt  = (bf16*)smem;                    // 256*264*2 = 135168
  bf16*  hbf = (bf16*)(smem + 135168);         // 256 bf16 (+pad)
  float* cst = (float*)(smem + 135168 + 512);  // 256 f32

  const int tid  = threadIdx.x;
  const int lane = tid & 31;
  const int wv   = tid >> 5;
  const int g    = blockIdx.x;                 // 0:i 1:f 2:g 3:o
  const int m    = lane & 15;

  if (tid < 256) { hbf[tid] = (bf16)0.f; cst[tid] = 0.f; }

#if HAVE_TDM
  float* Wscr = (float*)(smem + 135168 + 512 + 1024);  // 128*256*4 = 131072
  for (int hlf = 0; hlf < 2; ++hlf) {
    if (__builtin_amdgcn_readfirstlane(wv) == 0) {
      tdm_load_2d_f32(LDS_OFFSET(Wscr), Whh + (size_t)(g * 256 + hlf * 128) * 256,
                      128, 256, 256);
      __builtin_amdgcn_s_wait_tensorcnt((short)0);
    }
    __syncthreads();
    for (int e = tid; e < 128 * 256; e += 256) {
      int n = e >> 8, k = e & 255;
      Bt[(hlf * 128 + n) * BS_W + k] = (bf16)Wscr[e];
    }
    __syncthreads();
  }
#else
  for (int e = tid; e < 256 * 256; e += 256) {
    int n = e >> 8, k = e & 255;
    Bt[n * BS_W + k] = (bf16)Whh[(size_t)(g * 256 + n) * 256 + k];
  }
  __syncthreads();
#endif

  for (int t = 0; t < S_; ++t) {
    if (t + 1 < S_)  // pull next precomputed gate slice toward L2/L0
      __builtin_prefetch(&Xg[(size_t)(t + 1) * 1024 + g * 256 + tid], 0, 0);

    // y = h @ Whh_g.T via WMMA, only row 0 of the M=16 pad carries h.
    v8f acc[2] = { zero_v8f(), zero_v8f() };
#pragma unroll 2
    for (int kc = 0; kc < 8; ++kc) {
      int half = (lane >> 4) & 1;
      const v8bf* p0 = (const v8bf*)(hbf + kc * 32 + half * 8);
      const v8bf* p1 = (const v8bf*)(hbf + kc * 32 + 16 + half * 8);
      v8bf lo = *p0, hi = *p1;
      v16bf af;
#pragma unroll
      for (int i = 0; i < 8; ++i) {
        af[i]     = (m == 0) ? lo[i] : (bf16)0.f;
        af[8 + i] = (m == 0) ? hi[i] : (bf16)0.f;
      }
#pragma unroll
      for (int j = 0; j < 2; ++j) {
        v16bf bfr = frag16(&Bt[((wv * 2 + j) * 16 + m) * BS_W + kc * 32], lane);
        acc[j] = __builtin_amdgcn_wmma_f32_16x16x32_bf16(
            false, af, false, bfr, (short)0, acc[j], false, false);
      }
    }
    // Row 0 of each C tile lives in VGPR0 of lanes 0-15.
    if (lane < 16) {
#pragma unroll
      for (int j = 0; j < 2; ++j) {
        int n = (wv * 2 + j) * 16 + m;       // 0..255 within gate group
        float pre = acc[j][0] + Xg[(size_t)t * 1024 + g * 256 + n];
        float a = (g == 2) ? tanhf(pre) : sigmoidf_(pre);
        __hip_atomic_store(&act[(t & 1) * 1024 + g * 256 + n], a,
                           __ATOMIC_RELAXED, __HIP_MEMORY_SCOPE_AGENT);
      }
    }
    __threadfence();
    __syncthreads();
    if (tid == 0) {
      __hip_atomic_fetch_add(counter, 1u, __ATOMIC_RELEASE, __HIP_MEMORY_SCOPE_AGENT);
      unsigned target = 4u * (unsigned)(t + 1);
      while (__hip_atomic_load(counter, __ATOMIC_ACQUIRE, __HIP_MEMORY_SCOPE_AGENT) < target)
        __builtin_amdgcn_s_sleep(8);
    }
    __syncthreads();
    // Redundant c/h update in every block (keeps h local, no broadcast step).
    if (tid < 256) {
      const float* ab = act + (t & 1) * 1024;
      float ai  = __hip_atomic_load(&ab[tid],       __ATOMIC_RELAXED, __HIP_MEMORY_SCOPE_AGENT);
      float af_ = __hip_atomic_load(&ab[256 + tid], __ATOMIC_RELAXED, __HIP_MEMORY_SCOPE_AGENT);
      float ag  = __hip_atomic_load(&ab[512 + tid], __ATOMIC_RELAXED, __HIP_MEMORY_SCOPE_AGENT);
      float ao  = __hip_atomic_load(&ab[768 + tid], __ATOMIC_RELAXED, __HIP_MEMORY_SCOPE_AGENT);
      float c = af_ * cst[tid] + ai * ag;
      float h = ao * tanhf(c);
      cst[tid] = c;
      hbf[tid] = (bf16)h;
      if (g == 3) Hout[(size_t)t * 256 + tid] = h;
    }
    __syncthreads();
  }
}

// ---------------------------------------------------------------------------
// logits = H @ tag_W.T + tag_b, then log_softmax. One 64-thread block per word.
__global__ __launch_bounds__(64) void k_tag(
    const float* __restrict__ Hout, const float* __restrict__ tagW,
    const float* __restrict__ tagb, float* __restrict__ out) {
  __shared__ float hrow[256];
  __shared__ float red[64];
  const int w = blockIdx.x, tid = threadIdx.x;
  for (int k = tid; k < 256; k += 64) hrow[k] = Hout[(size_t)w * 256 + k];
  __syncthreads();
  float logit = -INFINITY;
  if (tid < TAGS_) {
    float s = tagb[tid];
#pragma unroll 4
    for (int k = 0; k < 256; ++k) s += hrow[k] * tagW[tid * 256 + k];
    logit = s;
  }
  red[tid] = logit; __syncthreads();
  for (int off = 32; off > 0; off >>= 1) {
    if (tid < off) red[tid] = fmaxf(red[tid], red[tid + off]);
    __syncthreads();
  }
  float mx = red[0]; __syncthreads();
  red[tid] = (tid < TAGS_) ? __expf(logit - mx) : 0.f; __syncthreads();
  for (int off = 32; off > 0; off >>= 1) {
    if (tid < off) red[tid] += red[tid + off];
    __syncthreads();
  }
  float lse = mx + __logf(red[0]);
  if (tid < TAGS_) out[(size_t)w * TAGS_ + tid] = logit - lse;
}

// ---------------------------------------------------------------------------
extern "C" void kernel_launch(void* const* d_in, const int* in_sizes, int n_in,
                              void* d_out, int out_size, void* d_ws, size_t ws_size,
                              hipStream_t stream) {
  (void)in_sizes; (void)n_in; (void)out_size; (void)ws_size;
  const int*   word_idxs = (const int*)d_in[0];
  const int*   char_idxs = (const int*)d_in[1];
  const int*   char_lens = (const int*)d_in[2];
  const float* char_emb  = (const float*)d_in[3];
  const float* cWih      = (const float*)d_in[4];
  const float* cWhh      = (const float*)d_in[5];
  const float* cbih      = (const float*)d_in[6];
  const float* cbhh      = (const float*)d_in[7];
  const float* word_emb  = (const float*)d_in[8];
  const float* wWih      = (const float*)d_in[9];
  const float* wWhh      = (const float*)d_in[10];
  const float* wbih      = (const float*)d_in[11];
  const float* wbhh      = (const float*)d_in[12];
  const float* tagW      = (const float*)d_in[13];
  const float* tagb      = (const float*)d_in[14];

  char* ws = (char*)d_ws;
  float*    char_feat = (float*)ws;                                    // 2 MB
  float*    Xg        = (float*)(ws + (size_t)2  * 1024 * 1024);       // 32 MB
  float*    Hout      = (float*)(ws + (size_t)34 * 1024 * 1024);       // 8 MB
  float*    act       = (float*)(ws + (size_t)42 * 1024 * 1024);       // 8 KB
  unsigned* counter   = (unsigned*)(ws + (size_t)42 * 1024 * 1024 + 8192);

  k_init<<<1, 1, 0, stream>>>(counter);

  size_t smemA = 69632 + 17408 + 65536 + 1024 + 16384 + 16384;  // ~182 KB
  k_char_lstm<<<S_ / 64, 256, smemA, stream>>>(
      char_idxs, char_lens, char_emb, cWih, cWhh, cbih, cbhh, char_feat);

  size_t smemB = 25600 + 25600 + 49152;                         // ~98 KB
  k_wih_gemm<<<128 * 16, 256, smemB, stream>>>(
      word_idxs, word_emb, char_feat, wWih, wbih, wbhh, Xg);

  size_t smemC = 135168 + 512 + 1024 + 131072;                  // ~262 KB
  k_word_lstm<<<4, 256, smemC, stream>>>(wWhh, Xg, act, counter, Hout);

  k_tag<<<S_, 64, 0, stream>>>(Hout, tagW, tagb, (float*)d_out);
}
